// PEERLora_1726576855274
// MI455X (gfx1250) — matrix operand order repaired and checked by reading, not compiled, not run on hardware
//
#include <hip/hip_runtime.h>
#include <math.h>

// ---------------------------------------------------------------------------
// Types for CDNA5 WMMA (wave32, 16x16x32 bf16 -> f32)
// ---------------------------------------------------------------------------
typedef __attribute__((ext_vector_type(8)))  __bf16 v8bf;
typedef __attribute__((ext_vector_type(16))) __bf16 v16bf;
typedef __attribute__((ext_vector_type(8)))  float  v8f;
typedef __attribute__((ext_vector_type(4)))  float  v4f;

#if __has_builtin(__builtin_amdgcn_global_load_async_to_lds_b128) && \
    __has_builtin(__builtin_amdgcn_s_wait_asynccnt)
#define USE_ASYNC_LDS 1
#else
#define USE_ASYNC_LDS 0
#endif

__device__ __forceinline__ unsigned short f2bf(float f) {
    unsigned int u = __builtin_bit_cast(unsigned int, f);
    u += 0x7FFFu + ((u >> 16) & 1u);            // round-to-nearest-even
    return (unsigned short)(u >> 16);
}

#if USE_ASYNC_LDS
// The builtin's parameters are int4 pointers: (v4i AS(1)* global, v4i AS(3)* lds,
// imm offset, imm cpol).  Build the address-space pointers via integer casts:
// generic LDS pointers carry the LDS byte offset in their low 32 bits; global
// generic pointers are the full 64-bit address.
typedef int v4si __attribute__((vector_size(16)));
typedef __attribute__((address_space(1))) v4si as1_v4si;
typedef __attribute__((address_space(3))) v4si as3_v4si;

__device__ __forceinline__ void async_copy16(const float* gsrc, float* lds_dst) {
    as1_v4si* g = (as1_v4si*)(unsigned long long)(uintptr_t)gsrc;
    as3_v4si* l = (as3_v4si*)(unsigned int)(uintptr_t)lds_dst;
    __builtin_amdgcn_global_load_async_to_lds_b128(g, l, 0, 0);
}
#endif

// ---------------------------------------------------------------------------
// Elementwise f32 -> bf16 convert
// ---------------------------------------------------------------------------
__global__ void cvt_bf16(const float* __restrict__ in, unsigned short* __restrict__ out, int n) {
    int i = blockIdx.x * blockDim.x + threadIdx.x;
    if (i < n) out[i] = f2bf(in[i]);
}

// Wq is (out=8192, in=1024) row-major; reduction dim for K_eff is Wq's row index.
// Produce WqT[p,h,d,dk] = Wq[(p*4096 + h*512 + dk), d] as bf16 (contiguous in dk).
__global__ void wq_transpose(const float* __restrict__ Wq, unsigned short* __restrict__ out, int total) {
    int i = blockIdx.x * blockDim.x + threadIdx.x;
    if (i >= total) return;
    int dk   = i & 511;
    int rest = i >> 9;
    int d    = rest & 1023;
    int ph   = rest >> 10;              // p*8 + h
    int p = ph >> 3, hh = ph & 7;
    int o = p * 4096 + hh * 512 + dk;   // Wq row
    out[i] = f2bf(Wq[(size_t)o * 1024 + d]);
}

// ---------------------------------------------------------------------------
// Generic bf16 WMMA GEMM: C[M,N] = A[M,K] * B[N,K]^T  (both row-major over K).
// One wave computes a 16(M) x 64(N) tile; block = 4 waves -> 16 x 256.
// Batched over blockIdx.z = p*hcount + h with per-(p,h) element offsets.
// ---------------------------------------------------------------------------
__global__ __launch_bounds__(128) void gemm_bf16_wmma(
    const unsigned short* __restrict__ A, const unsigned short* __restrict__ B,
    void* __restrict__ C, int M, int N, int K,
    int lda, int ldb, int ldc,
    int sAp, int sAh, int sBp, int sBh, long long sCp, long long sCh,
    int hcount, int bf16out)
{
    int z = blockIdx.z;
    int p = z / hcount;
    int h = z % hcount;
    const unsigned short* Ab = A + (size_t)p * sAp + (size_t)h * sAh;
    const unsigned short* Bb = B + (size_t)p * sBp + (size_t)h * sBh;
    size_t cOff = (size_t)(p * sCp + h * sCh);

    int wave = threadIdx.x >> 5;
    int lane = threadIdx.x & 31;
    int m0 = blockIdx.x * 16;
    int n0 = blockIdx.y * 256 + wave * 64;
    if (m0 >= M || n0 >= N) return;

    int arow  = lane & 15;
    int aksel = (lane >> 4) << 4;        // A: lanes 16-31 hold K+16..31
    int bksel = (lane >> 4) << 4;        // B: lanes 16-31 hold K+16..31
    int bcol  = lane & 15;

    const unsigned short* Aptr  = Ab + (size_t)(m0 + arow) * lda + aksel;
    const unsigned short* Bptr0 = Bb + (size_t)(n0 +  0 + bcol) * ldb + bksel;
    const unsigned short* Bptr1 = Bb + (size_t)(n0 + 16 + bcol) * ldb + bksel;
    const unsigned short* Bptr2 = Bb + (size_t)(n0 + 32 + bcol) * ldb + bksel;
    const unsigned short* Bptr3 = Bb + (size_t)(n0 + 48 + bcol) * ldb + bksel;

    v8f acc0 = {}, acc1 = {}, acc2 = {}, acc3 = {};

    #pragma unroll 4
    for (int k0 = 0; k0 < K; k0 += 32) {
        v16bf a = *(const v16bf*)(const void*)(Aptr + k0);
        v16bf b0 = *(const v16bf*)(const void*)(Bptr0 + k0);
        v16bf b1 = *(const v16bf*)(const void*)(Bptr1 + k0);
        v16bf b2 = *(const v16bf*)(const void*)(Bptr2 + k0);
        v16bf b3 = *(const v16bf*)(const void*)(Bptr3 + k0);
        acc0 = __builtin_amdgcn_wmma_f32_16x16x32_bf16(false, a, false, b0, (short)0, acc0, false, false);
        acc1 = __builtin_amdgcn_wmma_f32_16x16x32_bf16(false, a, false, b1, (short)0, acc1, false, false);
        acc2 = __builtin_amdgcn_wmma_f32_16x16x32_bf16(false, a, false, b2, (short)0, acc2, false, false);
        acc3 = __builtin_amdgcn_wmma_f32_16x16x32_bf16(false, a, false, b3, (short)0, acc3, false, false);
    }

    // C/D layout: VGPR r -> M = m0 + r + 8*(lane>>4), N = n0 + (lane&15) + 16*t
    int crow = m0 + ((lane >> 4) << 3);
    int ccol = lane & 15;
    float* Cf = (float*)C;
    unsigned short* Ch = (unsigned short*)C;
    #pragma unroll
    for (int r = 0; r < 8; ++r) {
        size_t base = cOff + (size_t)(crow + r) * ldc + n0 + ccol;
        if (bf16out) {
            Ch[base +  0] = f2bf(acc0[r]);
            Ch[base + 16] = f2bf(acc1[r]);
            Ch[base + 32] = f2bf(acc2[r]);
            Ch[base + 48] = f2bf(acc3[r]);
        } else {
            Cf[base +  0] = acc0[r];
            Cf[base + 16] = acc1[r];
            Cf[base + 32] = acc2[r];
            Cf[base + 48] = acc3[r];
        }
    }
}

// ---------------------------------------------------------------------------
// Top-1 selection. With TOPK=1 the best outer-sum is max(sim0)+max(sim1).
// sim layout: sim[token, p*1024 + h*128 + k], token = b*N + n.
// ---------------------------------------------------------------------------
__global__ void topk1_kernel(const float* __restrict__ sim, float* __restrict__ gate,
                             int* __restrict__ eidx, int total)
{
    int t = blockIdx.x * blockDim.x + threadIdx.x;
    if (t >= total) return;
    int token = t >> 3, hh = t & 7;
    const float* s0 = sim + (size_t)token * 2048 + hh * 128;
    const float* s1 = s0 + 1024;
    float m0 = s0[0], m1 = s1[0];
    int i0 = 0, i1 = 0;
    for (int j = 1; j < 128; ++j) {
        float v = s0[j]; if (v > m0) { m0 = v; i0 = j; }
        float w = s1[j]; if (w > m1) { m1 = w; i1 = j; }
    }
    float sc = m0 + m1;
    gate[t] = sc > 0.f ? sc : 0.f;   // ReLU gating
    eidx[t] = i0 * 128 + i1;
}

// ---------------------------------------------------------------------------
// Per-token expert input path: hidden += sum_h gate_h * (x . in_a[e_h]) * in_b[e_h]
// then exact GeLU; writes fp32 hidden (for the out-path dots) and bf16 copy
// (for the W_out WMMA GEMM). Block = 256 threads = 8 waves, 1 wave per head.
// The 8 in_b rows (64 KB) are staged into LDS with async global->LDS copies
// that overlap the dot-product phase; ASYNCcnt is drained before phase 2.
// ---------------------------------------------------------------------------
__global__ __launch_bounds__(256) void expert_in_kernel(
    const float* __restrict__ x, const float* __restrict__ in_a,
    const float* __restrict__ in_b, const float* __restrict__ gate,
    const int* __restrict__ eidx, float* __restrict__ hidden,
    unsigned short* __restrict__ hidbf)
{
    int token = blockIdx.x;
    extern __shared__ float rows[];          // 8 * 2048 floats (dynamic, 64 KB)
    __shared__ float lin[8];
    __shared__ int   sidx[8];
    int tid = threadIdx.x, wave = tid >> 5, lane = tid & 31;

    int e = eidx[token * 8 + wave];
    const float* brow = in_b + (size_t)e * 2048;
    float* lrow = rows + wave * 2048;
#if USE_ASYNC_LDS
    for (int j = lane * 4; j < 2048; j += 128)
        async_copy16(brow + j, lrow + j);
#else
    for (int j = lane * 4; j < 2048; j += 128)
        *(v4f*)(lrow + j) = *(const v4f*)(brow + j);
#endif

    const float* xa = x + (size_t)token * 1024;
    const float* Ae = in_a + (size_t)e * 1024;
    float s = 0.f;
    for (int j = lane; j < 1024; j += 32) s += xa[j] * Ae[j];
    for (int off = 16; off; off >>= 1) s += __shfl_xor(s, off);
    if (lane == 0) { lin[wave] = gate[token * 8 + wave] * s; sidx[wave] = e; }
#if USE_ASYNC_LDS
    __builtin_amdgcn_s_wait_asynccnt(0);     // each wave drains its own copies
#endif
    __syncthreads();

    for (int d = tid; d < 2048; d += 256) {
        float acc = hidden[(size_t)token * 2048 + d];
        #pragma unroll
        for (int hh = 0; hh < 8; ++hh)
            acc += lin[hh] * rows[hh * 2048 + d];
        float g = 0.5f * acc * (1.f + erff(acc * 0.70710678118654752f));  // exact GeLU
        hidden[(size_t)token * 2048 + d] = g;
        hidbf [(size_t)token * 2048 + d] = f2bf(g);
    }
}

// Per-token expert output path: out += sum_h gate_h * (hidden . out_a[e_h]) * out_b[e_h]
__global__ __launch_bounds__(256) void expert_out_kernel(
    const float* __restrict__ hidden, const float* __restrict__ out_a,
    const float* __restrict__ out_b, const float* __restrict__ gate,
    const int* __restrict__ eidx, float* __restrict__ out)
{
    int token = blockIdx.x;
    extern __shared__ float rows[];          // 8 * 1024 floats (dynamic, 32 KB)
    __shared__ float lin[8];
    __shared__ int   sidx[8];
    int tid = threadIdx.x, wave = tid >> 5, lane = tid & 31;

    int e = eidx[token * 8 + wave];
    const float* brow = out_b + (size_t)e * 1024;
    float* lrow = rows + wave * 1024;
#if USE_ASYNC_LDS
    for (int j = lane * 4; j < 1024; j += 128)
        async_copy16(brow + j, lrow + j);
#else
    for (int j = lane * 4; j < 1024; j += 128)
        *(v4f*)(lrow + j) = *(const v4f*)(brow + j);
#endif

    const float* ha = hidden + (size_t)token * 2048;
    const float* Ae = out_a + (size_t)e * 2048;
    float s = 0.f;
    for (int j = lane; j < 2048; j += 32) s += ha[j] * Ae[j];
    for (int off = 16; off; off >>= 1) s += __shfl_xor(s, off);
    if (lane == 0) { lin[wave] = gate[token * 8 + wave] * s; sidx[wave] = e; }
#if USE_ASYNC_LDS
    __builtin_amdgcn_s_wait_asynccnt(0);
#endif
    __syncthreads();

    for (int d = tid; d < 1024; d += 256) {
        float acc = out[(size_t)token * 1024 + d];
        #pragma unroll
        for (int hh = 0; hh < 8; ++hh)
            acc += lin[hh] * rows[hh * 1024 + d];
        out[(size_t)token * 1024 + d] = acc;
    }
}

// ---------------------------------------------------------------------------
// Host-side orchestration
// ---------------------------------------------------------------------------
extern "C" void kernel_launch(void* const* d_in, const int* in_sizes, int n_in,
                              void* d_out, int out_size, void* d_ws, size_t ws_size,
                              hipStream_t stream) {
    const float* x     = (const float*)d_in[0];   // [4,2048,1024]
    const float* Wq    = (const float*)d_in[1];   // [8192,1024]
    const float* keys  = (const float*)d_in[2];   // [8,128,2,512]
    const float* W_in  = (const float*)d_in[3];   // [2048,1024]
    const float* W_out = (const float*)d_in[4];   // [1024,2048]
    const float* in_a  = (const float*)d_in[5];   // [16384,1024]
    const float* in_b  = (const float*)d_in[6];   // [16384,2048]
    const float* out_a = (const float*)d_in[7];   // [16384,2048]
    const float* out_b = (const float*)d_in[8];   // [16384,1024]
    float* out = (float*)d_out;                   // [4,2048,1024]

    const int T = 8192;                 // B*N tokens
    // Workspace carve-up (256B aligned)
    char* w = (char*)d_ws;
    auto alloc = [&](size_t bytes) -> void* {
        void* p = (void*)w;
        w += (bytes + 255) & ~(size_t)255;
        return p;
    };
    unsigned short* xbf    = (unsigned short*)alloc((size_t)T * 1024 * 2);       // x bf16
    unsigned short* keysbf = (unsigned short*)alloc((size_t)8 * 128 * 2 * 512 * 2);
    unsigned short* wqt    = (unsigned short*)alloc((size_t)16 * 1024 * 512 * 2); // WqT[p,h,d,dk]
    unsigned short* winbf  = (unsigned short*)alloc((size_t)2048 * 1024 * 2);
    unsigned short* woutbf = (unsigned short*)alloc((size_t)1024 * 2048 * 2);
    unsigned short* keff   = (unsigned short*)alloc((size_t)16 * 128 * 1024 * 2); // K_eff[p,h,k,d] bf16
    float*          sim    = (float*)alloc((size_t)T * 2048 * 4);                 // sim[token, p*1024+h*128+k]
    float*          hidden = (float*)alloc((size_t)T * 2048 * 4);
    unsigned short* hidbf  = (unsigned short*)alloc((size_t)T * 2048 * 2);
    float*          gate   = (float*)alloc((size_t)T * 8 * 4);
    int*            eidx   = (int*)alloc((size_t)T * 8 * 4);

    // 1) Precision staging (fp32 -> bf16), Wq transposed so K is contiguous
    cvt_bf16<<<(8388608 + 255) / 256, 256, 0, stream>>>(x, xbf, 8388608);
    cvt_bf16<<<(1048576 + 255) / 256, 256, 0, stream>>>(keys, keysbf, 1048576);
    wq_transpose<<<(8388608 + 255) / 256, 256, 0, stream>>>(Wq, wqt, 8388608);
    cvt_bf16<<<(2097152 + 255) / 256, 256, 0, stream>>>(W_in, winbf, 2097152);
    cvt_bf16<<<(2097152 + 255) / 256, 256, 0, stream>>>(W_out, woutbf, 2097152);

    // 2) Fold Wq into keys: K_eff[p,h] (128x1024) = keys[h,:,p,:] (128x512, row stride 1024)
    //    x WqT[p,h] (1024x512)^T.  Batched over z = p*8+h, bf16 output.
    gemm_bf16_wmma<<<dim3(8, 4, 16), 128, 0, stream>>>(
        keysbf, wqt, keff, 128, 1024, 512,
        /*lda*/1024, /*ldb*/512, /*ldc*/1024,
        /*sAp*/512, /*sAh*/131072, /*sBp*/4194304, /*sBh*/524288,
        /*sCp*/1048576LL, /*sCh*/131072LL, /*hcount*/8, /*bf16out*/1);

    // 3) sim = x @ K_eff^T : [8192 x 2048] over K=1024, fp32 out
    gemm_bf16_wmma<<<dim3(512, 8, 1), 128, 0, stream>>>(
        xbf, keff, sim, 8192, 2048, 1024,
        1024, 1024, 2048, 0, 0, 0, 0, 0LL, 0LL, 1, 0);

    // 4) top-1 expert per (token, head): argmax(sim0)+argmax(sim1), ReLU gate
    topk1_kernel<<<(65536 + 255) / 256, 256, 0, stream>>>(sim, gate, eidx, 65536);

    // 5) dense hidden = x @ W_in^T : [8192 x 2048] over K=1024
    gemm_bf16_wmma<<<dim3(512, 8, 1), 128, 0, stream>>>(
        xbf, winbf, hidden, 8192, 2048, 1024,
        1024, 1024, 2048, 0, 0, 0, 0, 0LL, 0LL, 1, 0);

    // 6) expert rank-1 input updates + exact GeLU (writes fp32 + bf16 hidden)
    expert_in_kernel<<<T, 256, 8 * 2048 * sizeof(float), stream>>>(
        x, in_a, in_b, gate, eidx, hidden, hidbf);

    // 7) dense out = gelu(hidden) @ W_out^T : [8192 x 1024] over K=2048 (overwrites d_out)
    gemm_bf16_wmma<<<dim3(512, 4, 1), 128, 0, stream>>>(
        hidbf, woutbf, out, 8192, 1024, 2048,
        2048, 2048, 1024, 0, 0, 0, 0, 0LL, 0LL, 1, 0);

    // 8) expert rank-1 output updates (in-place add into d_out)
    expert_out_kernel<<<T, 256, 8 * 1024 * sizeof(float), stream>>>(
        hidden, out_a, out_b, gate, eidx, out);
}
